// Model_10582799417658
// MI455X (gfx1250) — matrix operand order repaired and checked by reading, not compiled, not run on hardware
//
#include <hip/hip_runtime.h>

typedef __attribute__((ext_vector_type(16))) _Float16 v16h;
typedef __attribute__((ext_vector_type(8)))  _Float16 v8h;
typedef __attribute__((ext_vector_type(8)))  float    v8f;

#define BATCH 4096
#define SEQ   128
#define DIM   64
#define HID   256
#define LAT   64
#define ROWS  64
#define NTHREADS 512

#define WMMA_F16(a, b, c) \
  __builtin_amdgcn_wmma_f32_16x16x32_f16(false, (a), false, (b), (short)0, (c), false, false)

// Scheduler fence: nothing may be reordered across.
#define SCHED_FENCE() __builtin_amdgcn_sched_barrier(0)

// Build 16x32 f16 A-fragment from row-major f16 tile (contiguous K order):
// lane&15 = row M; lanes>=16 get K base +16; elements 0..15 are K base+0..15.
__device__ __forceinline__ v16h load_A(const _Float16* base, int ld, int lane) {
    int row = lane & 15;
    int kb  = (lane >> 4) << 4;           // 0 or 16
    const _Float16* p = base + row * ld + kb;
    v8h lo = *(const v8h*)(p);            // K = kb .. kb+7
    v8h hi = *(const v8h*)(p + 8);        // K = kb+8 .. kb+15
    return __builtin_shufflevector(lo, hi, 0, 1, 2, 3, 4, 5, 6, 7,
                                           8, 9, 10, 11, 12, 13, 14, 15);
}

// Build 32x16 f16 B-fragment from row-major weight W[N][K] (out = in @ W^T):
// lane&15 = col N; lanes 0-15 hold K=k0..k0+15, lanes 16-31 hold K=k0+16..k0+31.
// 32-bit element offset off a global-AS base -> global_load_b128 (saddr+voffset).
__device__ __forceinline__ v16h load_B(const _Float16* w, int ld,
                                       int n0, int k0, int lane) {
    int n  = n0 + (lane & 15);
    int kb = k0 + ((lane >> 4) << 4);     // 0 or 16
    return *(const v16h*)(w + (n * ld + kb));
}

__device__ __forceinline__ float sigmoidf_(float x) {
    return 1.0f / (1.0f + __expf(-x));
}

__global__ void f32_to_f16_kernel(const float* __restrict__ src,
                                  _Float16* __restrict__ dst, int n) {
    int i = blockIdx.x * blockDim.x + threadIdx.x;
    if (i < n) dst[i] = (_Float16)src[i];
}

__global__ __launch_bounds__(NTHREADS, 1)
void vae_lstm_kernel(const float* __restrict__ data,
                     const float* __restrict__ locs,
                     const _Float16* __restrict__ w_ih_g,   // [1024][64] f16
                     const _Float16* __restrict__ w_hh_g,   // [1024][256] f16
                     const float* __restrict__ b_ih, const float* __restrict__ b_hh,
                     const _Float16* __restrict__ w_head_g, // [256][256] f16 (mu|var|velo|switch)
                     const float* __restrict__ b_mu, const float* __restrict__ b_var,
                     const float* __restrict__ b_velo, const float* __restrict__ b_sw,
                     const float* __restrict__ w_gen, const float* __restrict__ b_gen,
                     const float* __restrict__ w_on,  const float* __restrict__ b_on,
                     const _Float16* __restrict__ w_next_g, // [64][64] f16
                     const float* __restrict__ b_next,
                     const float* __restrict__ w_loc, const float* __restrict__ b_loc,
                     const float* __restrict__ eps,         // [127][B][64]
                     float* __restrict__ out_states,        // [B][127][64]
                     float* __restrict__ out_gens,          // [B][127]
                     float* __restrict__ out_onoff,         // [B][127]
                     float* __restrict__ out_tng,           // [B][128]
                     float* __restrict__ out_tg)            // [B][128]
{
    __shared__ __attribute__((aligned(32))) _Float16 xbuf[ROWS * DIM];    // x_t (f16)
    __shared__ __attribute__((aligned(32))) _Float16 hbuf0[ROWS * HID];   // h ping
    __shared__ __attribute__((aligned(32))) _Float16 hbuf1[ROWS * HID];   // h pong
    __shared__ __attribute__((aligned(32))) _Float16 zbuf[ROWS * LAT];    // z (f16)
    __shared__ __attribute__((aligned(32))) float    vs0[ROWS * LAT];     // velo, then state
    __shared__ __attribute__((aligned(32))) float    vs1[ROWS * LAT];     // switch
    __shared__ __attribute__((aligned(32))) float    dbuf[ROWS * LAT];    // delta
    __shared__ float bbuf[4 * HID];                                       // b_ih + b_hh
    __shared__ float genArr[ROWS];
    __shared__ float onArr[ROWS];

    const int tid    = threadIdx.x;
    const int wave   = tid >> 5;          // wave32
    const int lane   = tid & 31;
    const int wgBase = blockIdx.x * ROWS;
    const int mTile  = wave & 3;          // 4 row-tiles of 16
    const int rr     = wave >> 2;         // column residue class 0..3
    const int hi8    = (lane >> 4) << 3;  // C/D layout: +8 rows for upper half-wave
    const int ncol   = lane & 15;

    // ---- init: h0 = locs @ W_loc^T + b_loc, x0 = data[:,0,:], fused biases ----
    for (int i = tid; i < ROWS * HID; i += NTHREADS) {
        int row = i >> 8, j = i & 255;
        int b = wgBase + row;
        float v = locs[b * 2 + 0] * w_loc[j * 2 + 0]
                + locs[b * 2 + 1] * w_loc[j * 2 + 1] + b_loc[j];
        hbuf0[i] = (_Float16)v;
    }
    for (int i = tid; i < ROWS * DIM; i += NTHREADS) {
        int row = i >> 6, col = i & 63;
        xbuf[i] = (_Float16)data[(size_t)(wgBase + row) * SEQ * DIM + col];
    }
    for (int i = tid; i < 4 * HID; i += NTHREADS) bbuf[i] = b_ih[i] + b_hh[i];
    if (tid < ROWS) {  // leading zero column of totals
        out_tng[(size_t)(wgBase + tid) * SEQ] = 0.0f;
        out_tg [(size_t)(wgBase + tid) * SEQ] = 0.0f;
    }

    v8f c_reg[4];
#pragma unroll
    for (int cb = 0; cb < 4; ++cb)
#pragma unroll
        for (int r = 0; r < 8; ++r) c_reg[cb][r] = 0.0f;

    __syncthreads();

    for (int t = 1; t < SEQ; ++t) {
        const _Float16* hrd = (t & 1) ? hbuf0 : hbuf1;   // h_{t-1}
        _Float16*       hwr = (t & 1) ? hbuf1 : hbuf0;   // h_t

        // Defeat LICM without destroying pointer provenance: a laundered
        // zero offset is "redefined" every iteration, so the B-fragment
        // loads cannot be hoisted out of the time loop, yet the bases keep
        // their global address space -> global_load_b128 with SGPR base.
        size_t zoff = 0;
        asm volatile("" : "+s"(zoff));
        const _Float16* w_ih   = w_ih_g   + zoff;
        const _Float16* w_hh   = w_hh_g   + zoff;
        const _Float16* w_head = w_head_g + zoff;
        const _Float16* w_next = w_next_g + zoff;

        // ================= LSTM gates: 64x1024 via WMMA =================
#pragma unroll
        for (int cb = 0; cb < 4; ++cb) {
            const int ntI = rr + 4 * cb;          // i-gate column tile (0..15)
            v8f accI = {}, accF = {}, accG = {}, accO = {};
#pragma unroll
            for (int kt = 0; kt < 2; ++kt) {      // x @ W_ih^T, K = 64
                v16h a  = load_A(xbuf + mTile * 16 * DIM + kt * 32, DIM, lane);
                v16h bi = load_B(w_ih, DIM, (ntI +  0) * 16, kt * 32, lane);
                v16h bf = load_B(w_ih, DIM, (ntI + 16) * 16, kt * 32, lane);
                v16h bg = load_B(w_ih, DIM, (ntI + 32) * 16, kt * 32, lane);
                v16h bo = load_B(w_ih, DIM, (ntI + 48) * 16, kt * 32, lane);
                accI = WMMA_F16(a, bi, accI);
                accF = WMMA_F16(a, bf, accF);
                accG = WMMA_F16(a, bg, accG);
                accO = WMMA_F16(a, bo, accO);
                SCHED_FENCE();
            }
            // h @ W_hh^T, K = 256 : one-deep software pipeline on B-fragments
            {
                v16h pbi[2], pbf[2], pbg[2], pbo[2];
                pbi[0] = load_B(w_hh, HID, (ntI +  0) * 16, 0, lane);
                pbf[0] = load_B(w_hh, HID, (ntI + 16) * 16, 0, lane);
                pbg[0] = load_B(w_hh, HID, (ntI + 32) * 16, 0, lane);
                pbo[0] = load_B(w_hh, HID, (ntI + 48) * 16, 0, lane);
#pragma unroll
                for (int kt = 0; kt < 8; ++kt) {
                    const int cur = kt & 1, nxt = cur ^ 1;
                    if (kt < 7) {                 // prefetch next K-tile
                        pbi[nxt] = load_B(w_hh, HID, (ntI +  0) * 16, (kt + 1) * 32, lane);
                        pbf[nxt] = load_B(w_hh, HID, (ntI + 16) * 16, (kt + 1) * 32, lane);
                        pbg[nxt] = load_B(w_hh, HID, (ntI + 32) * 16, (kt + 1) * 32, lane);
                        pbo[nxt] = load_B(w_hh, HID, (ntI + 48) * 16, (kt + 1) * 32, lane);
                    }
                    v16h a = load_A(hrd + mTile * 16 * HID + kt * 32, HID, lane);
                    accI = WMMA_F16(a, pbi[cur], accI);
                    accF = WMMA_F16(a, pbf[cur], accF);
                    accG = WMMA_F16(a, pbg[cur], accG);
                    accO = WMMA_F16(a, pbo[cur], accO);
                    SCHED_FENCE();
                }
            }
            const int nI = ntI * 16 + ncol;       // gate-internal column 0..255
            const float bi = bbuf[nI      ];
            const float bf = bbuf[nI + 256];
            const float bg = bbuf[nI + 512];
            const float bo = bbuf[nI + 768];
#pragma unroll
            for (int r = 0; r < 8; ++r) {
                float gi = sigmoidf_(accI[r] + bi);
                float gf = sigmoidf_(accF[r] + bf);
                float gg = tanhf    (accG[r] + bg);
                float go = sigmoidf_(accO[r] + bo);
                float cn = gf * c_reg[cb][r] + gi * gg;
                c_reg[cb][r] = cn;                // cell state stays in VGPRs
                // write h_t tile straight into the pong buffer (no conflict
                // with readers of hrd, so no barrier / register array needed)
                hwr[(mTile * 16 + r + hi8) * HID + nI] = (_Float16)(go * tanhf(cn));
            }
            SCHED_FENCE();
        }
        __syncthreads();                          // h_t complete in hwr

        // ============ 4 heads: 64x256 = h(64x256) @ W_head^T ============
        v8f hacc[4] = {{}, {}, {}, {}};
#pragma unroll
        for (int kt = 0; kt < 8; ++kt) {
            v16h a = load_A(hwr + mTile * 16 * HID + kt * 32, HID, lane);
#pragma unroll
            for (int hd = 0; hd < 4; ++hd) {
                v16h b = load_B(w_head, HID, (hd * 4 + rr) * 16, kt * 32, lane);
                hacc[hd] = WMMA_F16(a, b, hacc[hd]);
            }
            SCHED_FENCE();
        }
        {
            const int l = rr * 16 + ncol;         // latent index 0..63
            const float bm = b_mu[l], bv = b_var[l], bl = b_velo[l], bs = b_sw[l];
#pragma unroll
            for (int r = 0; r < 8; ++r) {
                int row = mTile * 16 + r + hi8;
                int b   = wgBase + row;
                float mu = hacc[0][r] + bm;
                float lv = hacc[1][r] + bv;
                float e  = eps[((size_t)(t - 1) * BATCH + b) * LAT + l];
                zbuf[row * LAT + l] = (_Float16)(e * __expf(0.5f * lv) + mu);
                vs0 [row * LAT + l] = hacc[2][r] + bl;   // velo
                vs1 [row * LAT + l] = hacc[3][r] + bs;   // switch
            }
        }
        SCHED_FENCE();
        __syncthreads();

        // ============ delta: 64x64 = z(64x64) @ W_next^T ============
        {
            v8f dacc = {};
#pragma unroll
            for (int kt = 0; kt < 2; ++kt) {
                v16h a = load_A(zbuf + mTile * 16 * LAT + kt * 32, LAT, lane);
                v16h b = load_B(w_next, LAT, rr * 16, kt * 32, lane);
                dacc = WMMA_F16(a, b, dacc);
            }
            int col = rr * 16 + ncol;
            float bn = b_next[col];
#pragma unroll
            for (int r = 0; r < 8; ++r)
                dbuf[(mTile * 16 + r + hi8) * LAT + col] = dacc[r] + bn;
        }
        SCHED_FENCE();
        __syncthreads();

        // ===== per-row scalars: generate = relu(velo.w_gen), on_off =====
        if (tid < ROWS) {
            float sg = 0.0f, so = 0.0f;
#pragma unroll 8
            for (int k = 0; k < LAT; ++k) {
                sg += vs0[tid * LAT + k] * w_gen[k];
                so += vs1[tid * LAT + k] * w_on[k];
            }
            float gen = fmaxf(sg + b_gen[0], 0.0f);
            float onf = (so + b_on[0] > 0.0f) ? 1.0f : 0.0f;
            genArr[tid] = gen;
            onArr[tid]  = onf;
            size_t bb = wgBase + tid;
            out_gens [bb * 127 + (t - 1)] = gen;
            out_onoff[bb * 127 + (t - 1)] = onf;
        }
        __syncthreads();                          // velo (vs0) now dead

        // ===== state = relu([0, x[:,1:]+delta[:,1:]]) into vs0 =====
        {
            int row = tid >> 3, c0 = (tid & 7) * 8;
#pragma unroll
            for (int j = 0; j < 8; ++j) {
                int col = c0 + j;
                float s = (col == 0) ? 0.0f
                        : fmaxf((float)xbuf[row * DIM + col] + dbuf[row * LAT + col], 0.0f);
                vs0[row * DIM + col] = s;
            }
        }
        __syncthreads();
        if (tid < ROWS) {                         // row totals + generate add
            float tp = 0.0f;
            for (int k = 0; k < DIM; ++k) tp += vs0[tid * DIM + k];
            float add = genArr[tid] * onArr[tid];
            vs0[tid * DIM + 0] = add;             // state[:,0] += add (was 0)
            size_t bb = wgBase + tid;
            out_tng[bb * SEQ + t] = tp;
            out_tg [bb * SEQ + t] = tp + add;
        }
        __syncthreads();
        {   // write states output; state becomes next x
            int row = tid >> 3, c0 = (tid & 7) * 8;
            size_t bb = wgBase + row;
#pragma unroll
            for (int j = 0; j < 8; ++j) {
                int col = c0 + j;
                float s = vs0[row * DIM + col];
                out_states[(bb * 127 + (t - 1)) * (size_t)DIM + col] = s;
                xbuf[row * DIM + col] = (_Float16)s;
            }
        }
        __syncthreads();
    }
}

extern "C" void kernel_launch(void* const* d_in, const int* in_sizes, int n_in,
                              void* d_out, int out_size, void* d_ws, size_t ws_size,
                              hipStream_t stream) {
    (void)in_sizes; (void)n_in; (void)out_size; (void)ws_size;

    const float* data   = (const float*)d_in[0];
    const float* locs   = (const float*)d_in[1];
    const float* W_ih   = (const float*)d_in[2];
    const float* W_hh   = (const float*)d_in[3];
    const float* b_ih   = (const float*)d_in[4];
    const float* b_hh   = (const float*)d_in[5];
    const float* W_mu   = (const float*)d_in[6];
    const float* b_mu   = (const float*)d_in[7];
    const float* W_var  = (const float*)d_in[8];
    const float* b_var  = (const float*)d_in[9];
    const float* W_velo = (const float*)d_in[10];
    const float* b_velo = (const float*)d_in[11];
    const float* W_sw   = (const float*)d_in[12];
    const float* b_sw   = (const float*)d_in[13];
    const float* W_gen  = (const float*)d_in[14];
    const float* b_gen  = (const float*)d_in[15];
    const float* W_on   = (const float*)d_in[16];
    const float* b_on   = (const float*)d_in[17];
    const float* W_next = (const float*)d_in[18];
    const float* b_next = (const float*)d_in[19];
    const float* W_loc  = (const float*)d_in[20];
    const float* b_loc  = (const float*)d_in[21];
    const float* eps    = (const float*)d_in[22];

    // f16 weight staging in workspace
    _Float16* ws      = (_Float16*)d_ws;
    _Float16* w_ih_h  = ws;                         // 1024*64
    _Float16* w_hh_h  = w_ih_h + 65536;             // 1024*256
    _Float16* w_hd_h  = w_hh_h + 262144;            // 256*256 (mu|var|velo|switch)
    _Float16* w_nx_h  = w_hd_h + 65536;             // 64*64

    f32_to_f16_kernel<<<(65536  + 255) / 256, 256, 0, stream>>>(W_ih,   w_ih_h, 65536);
    f32_to_f16_kernel<<<(262144 + 255) / 256, 256, 0, stream>>>(W_hh,   w_hh_h, 262144);
    f32_to_f16_kernel<<<64, 256, 0, stream>>>(W_mu,   w_hd_h,          16384);
    f32_to_f16_kernel<<<64, 256, 0, stream>>>(W_var,  w_hd_h + 16384,  16384);
    f32_to_f16_kernel<<<64, 256, 0, stream>>>(W_velo, w_hd_h + 32768,  16384);
    f32_to_f16_kernel<<<64, 256, 0, stream>>>(W_sw,   w_hd_h + 49152,  16384);
    f32_to_f16_kernel<<<16, 256, 0, stream>>>(W_next, w_nx_h,          4096);

    float* out        = (float*)d_out;
    float* out_states = out;                                        // B*127*64
    float* out_gens   = out_states + (size_t)BATCH * 127 * 64;      // B*127
    float* out_onoff  = out_gens   + (size_t)BATCH * 127;           // B*127
    float* out_tng    = out_onoff  + (size_t)BATCH * 127;           // B*128
    float* out_tg     = out_tng    + (size_t)BATCH * 128;           // B*128

    vae_lstm_kernel<<<BATCH / ROWS, NTHREADS, 0, stream>>>(
        data, locs, w_ih_h, w_hh_h, b_ih, b_hh,
        w_hd_h, b_mu, b_var, b_velo, b_sw,
        W_gen, b_gen, W_on, b_on,
        w_nx_h, b_next, W_loc, b_loc, eps,
        out_states, out_gens, out_onoff, out_tng, out_tg);
}
